// VectorQuantizer_14886356648663
// MI455X (gfx1250) — matrix operand order repaired and checked by reading, not compiled, not run on hardware
//
#include <hip/hip_runtime.h>
#include <hip/hip_bf16.h>

#define NPTS   65536          // 16*64*64
#define EDIM   129
#define SDIM   128
#define RBINS  16
#define ABINS  1024
#define NE     (RBINS*ABINS)  // 16384
#define TOPW   5
#define ZSIZE  (NPTS*EDIM)    // 8454144
#define NTILE  (ABINS/16)     // 64 bin tiles
#define WAVES_PER_BLK 8
#define PTS_PER_WAVE  32      // two 16-col WMMA groups per wave

typedef __attribute__((ext_vector_type(16))) _Float16 v16h;
typedef __attribute__((ext_vector_type(8)))  float    v8f;

union AFrag { v16h v; _Float16 h[16]; };
union CFrag { v8f  v; float    f[8];  };

// ---------------------------------------------------------------- init
__global__ void k_init(float* __restrict__ counts) {
    int i = blockIdx.x * blockDim.x + threadIdx.x;
    if (i < NE) counts[i] = 0.0f;
}

// ---------------------------------------------------------------- codebook f32 -> f16 (once)
__global__ void k_cvt_aw(const float* __restrict__ aw,
                         _Float16* __restrict__ awh) {
    int i = blockIdx.x * blockDim.x + threadIdx.x;
    if (i < ABINS * SDIM) awh[i] = (_Float16)aw[i];
}

// ---------------------------------------------------------------- prep
__global__ __launch_bounds__(256) void k_prep(const float* __restrict__ u,
                                              const float* __restrict__ rcen,
                                              float* __restrict__ invn_out,
                                              int*   __restrict__ topr_out) {
    int p = blockIdx.x * blockDim.x + threadIdx.x;
    if (p >= NPTS) return;
    const float* up = u + (size_t)p * EDIM;
    float u0 = up[0];
    float ss = 0.0f;
#pragma unroll 4
    for (int d = 0; d < SDIM; ++d) { float x = up[1 + d]; ss += x * x; }
    invn_out[p] = 1.0f / sqrtf(ss);
    float r = acoshf(fmaxf(u0, 1.0f + 0.01f));
    float b0 = 3.4e38f, b1 = 3.4e38f, b2 = 3.4e38f;
    int   i0 = 0, i1 = 0, i2 = 0;
#pragma unroll
    for (int j = 0; j < RBINS; ++j) {
        float rc = fminf(fmaxf(rcen[j], 0.01f), 1.1f);
        float d  = fabsf(r - rc);
        if      (d < b0) { b2 = b1; i2 = i1; b1 = b0; i1 = i0; b0 = d; i0 = j; }
        else if (d < b1) { b2 = b1; i2 = i1; b1 = d;  i1 = j; }
        else if (d < b2) { b2 = d;  i2 = j; }
    }
    topr_out[p] = i0 | (i1 << 5) | (i2 << 10);
}

// ---------------------------------------------------------------- top-5 insert
__device__ __forceinline__ void topk_insert(float tv[TOPW], int ti[TOPW],
                                            float v, int idx) {
    if (v > tv[TOPW - 1]) {
        tv[TOPW - 1] = v; ti[TOPW - 1] = idx;
#pragma unroll
        for (int s = TOPW - 1; s > 0; --s) {
            if (tv[s] > tv[s - 1]) {
                float fv = tv[s]; tv[s] = tv[s - 1]; tv[s - 1] = fv;
                int   fi = ti[s]; ti[s] = ti[s - 1]; ti[s - 1] = fi;
            }
        }
    }
}

// ---------------------------------------------------------------- WMMA matmul + top-5
// Block = 8 waves x 32 points = 256 points; each bin tile (16 bins x 128 dims, f16,
// 4 KB) is staged once per block in double-buffered LDS and reused by all 8 waves,
// each issuing 8 WMMAs per tile (two 16-point column groups share each A fragment).
__global__ __launch_bounds__(256) void k_sim(const float* __restrict__ u,
                                             const _Float16* __restrict__ awh,
                                             const float* __restrict__ invn,
                                             int* __restrict__ top5) {
    __shared__ alignas(16) _Float16 smA[2][16 * SDIM];   // 2 x 4 KB

    int tid  = threadIdx.x;
    int lane = tid & 31;
    int l    = lane & 15;
    int hi   = lane >> 4;
    int pbase = (blockIdx.x * WAVES_PER_BLK + (tid >> 5)) * PTS_PER_WAVE;

    // B fragments for the two 16-point groups of this wave
    AFrag bf0[4], bf1[4];
    {
        int p0 = pbase + l, p1 = pbase + 16 + l;
        float i0 = invn[p0], i1 = invn[p1];
        const float* us0 = u + (size_t)p0 * EDIM + 1;
        const float* us1 = u + (size_t)p1 * EDIM + 1;
#pragma unroll
        for (int ks = 0; ks < 4; ++ks) {
            int kb = ks * 32 + hi * 16;
#pragma unroll
            for (int e = 0; e < 16; ++e) {
                bf0[ks].h[e] = (_Float16)(us0[kb + e] * i0);
                bf1[ks].h[e] = (_Float16)(us1[kb + e] * i1);
            }
        }
    }

    float tv0[TOPW], tv1[TOPW]; int ti0[TOPW], ti1[TOPW];
#pragma unroll
    for (int k = 0; k < TOPW; ++k) {
        tv0[k] = -3.4e38f; ti0[k] = 0;
        tv1[k] = -3.4e38f; ti1[k] = 0;
    }

    // prolog: stage tile 0 (each thread moves 16 B; 256 threads cover 4 KB)
    {
        uint4 s0 = *(const uint4*)(awh + (size_t)tid * 8);
        *(uint4*)(&smA[0][tid * 8]) = s0;
    }
    __syncthreads();

    for (int t = 0; t < NTILE; ++t) {
        int cur = t & 1;
        uint4 nx;
        if (t + 1 < NTILE)
            nx = *(const uint4*)(awh + (size_t)(t + 1) * 16 * SDIM + tid * 8);

        const _Float16* as = &smA[cur][l * SDIM];   // A row = bin t*16 + l
        CFrag c0, c1;
        c0.v = (v8f){0.f, 0.f, 0.f, 0.f, 0.f, 0.f, 0.f, 0.f};
        c1.v = (v8f){0.f, 0.f, 0.f, 0.f, 0.f, 0.f, 0.f, 0.f};
#pragma unroll
        for (int ks = 0; ks < 4; ++ks) {
            AFrag a;
            int g1 = ks * 32 + hi * 16;             // K = {g1..g1+15}
#pragma unroll
            for (int e = 0; e < 16; ++e) {
                a.h[e]     = as[g1 + e];
            }
            c0.v = __builtin_amdgcn_wmma_f32_16x16x32_f16(
                false, a.v, false, bf0[ks].v, (short)0, c0.v, false, false);
            c1.v = __builtin_amdgcn_wmma_f32_16x16x32_f16(
                false, a.v, false, bf1[ks].v, (short)0, c1.v, false, false);
        }
#pragma unroll
        for (int j = 0; j < 8; ++j) {
            int idx = t * 16 + hi * 8 + j;
            topk_insert(tv0, ti0, c0.f[j], idx);
            topk_insert(tv1, ti1, c1.f[j], idx);
        }

        if (t + 1 < NTILE)
            *(uint4*)(&smA[cur ^ 1][tid * 8]) = nx;
        __syncthreads();
    }

    // merge the two half-lane lists (same point, disjoint bin rows)
    float sv[TOPW]; int si[TOPW];
#pragma unroll
    for (int k = 0; k < TOPW; ++k) { sv[k] = tv0[k]; si[k] = ti0[k]; }
#pragma unroll
    for (int k = 0; k < TOPW; ++k) {
        float ov = __shfl_xor(sv[k], 16, 32);
        int   oi = __shfl_xor(si[k], 16, 32);
        topk_insert(tv0, ti0, ov, oi);
    }
#pragma unroll
    for (int k = 0; k < TOPW; ++k) { sv[k] = tv1[k]; si[k] = ti1[k]; }
#pragma unroll
    for (int k = 0; k < TOPW; ++k) {
        float ov = __shfl_xor(sv[k], 16, 32);
        int   oi = __shfl_xor(si[k], 16, 32);
        topk_insert(tv1, ti1, ov, oi);
    }
    if (hi == 0) {
#pragma unroll
        for (int k = 0; k < TOPW; ++k) {
            top5[(size_t)(pbase + l) * TOPW + k]      = ti0[k];
            top5[(size_t)(pbase + 16 + l) * TOPW + k] = ti1[k];
        }
    }
}

// ---------------------------------------------------------------- assign + outputs
__global__ __launch_bounds__(256) void k_assign(const float* __restrict__ u,
                                                const float* __restrict__ aw,
                                                const float* __restrict__ rcen,
                                                const int*   __restrict__ topr,
                                                const int*   __restrict__ top5,
                                                float* __restrict__ out,
                                                float* __restrict__ tad_out,
                                                float* __restrict__ counts) {
    int lane = threadIdx.x & 31;
    int p    = blockIdx.x * 8 + (threadIdx.x >> 5);
    const float* up = u + (size_t)p * EDIM;
    float u0 = up[0];
    int d0 = lane * 4;
    float usl[4];
#pragma unroll
    for (int j = 0; j < 4; ++j) usl[j] = up[1 + d0 + j];

    int t5[TOPW];
#pragma unroll
    for (int k = 0; k < TOPW; ++k) t5[k] = top5[(size_t)p * TOPW + k];

    float dots[TOPW];
#pragma unroll
    for (int k = 0; k < TOPW; ++k) {
        const float* w = aw + (size_t)t5[k] * SDIM + d0;
        float s = usl[0]*w[0] + usl[1]*w[1] + usl[2]*w[2] + usl[3]*w[3];
#pragma unroll
        for (int off = 16; off > 0; off >>= 1) s += __shfl_down(s, off, 32);
        dots[k] = __shfl(s, 0, 32);
    }

    int tr = topr[p];
    int ridx[3] = { tr & 31, (tr >> 5) & 31, (tr >> 10) & 31 };
    float best = 3.4e38f; int bj = 0, bk = 0;
#pragma unroll
    for (int j = 0; j < 3; ++j) {
        float rc = fminf(fmaxf(rcen[ridx[j]], 0.01f), 1.1f);
        float ch = coshf(rc), sh = sinhf(rc);
#pragma unroll
        for (int k = 0; k < TOPW; ++k) {
            float inner = u0 * ch - sh * dots[k];
            float d = acoshf(fmaxf(inner, 1.0f + 1e-7f));
            if (d < best) { best = d; bj = j; bk = k; }
        }
    }
    int br = ridx[bj], bw = t5[bk];
    float rc = fminf(fmaxf(rcen[br], 0.01f), 1.1f);
    float sh = sinhf(rc);
    const float* wb = aw + (size_t)bw * SDIM + d0;
    float ssq = 0.0f;
    float* zrow = out + 1 + (size_t)p * EDIM;
#pragma unroll
    for (int j = 0; j < 4; ++j) {
        float x = sh * wb[j];
        zrow[1 + d0 + j] = x;
        ssq += x * x;
    }
#pragma unroll
    for (int off = 16; off > 0; off >>= 1) ssq += __shfl_down(ssq, off, 32);
    if (lane == 0) {
        float x0 = sqrtf(1.0f + ssq);        // projx time component
        zrow[0] = x0;
        float inner = u0 * x0 - sh * dots[bk];
        float ld = acoshf(fmaxf(inner, 1.0f + 1e-7f));
        float ru = acoshf(fmaxf(u0, 1.0f + 1e-5f));
        float rq = acoshf(fmaxf(x0, 1.0f + 1e-5f));
        tad_out[p] = ld + fabsf(ru - rq);
        atomicAdd(&counts[br * ABINS + bw], 1.0f);   // exact integer counts
    }
}

// ---------------------------------------------------------------- finalize
__global__ __launch_bounds__(256) void k_final(const float* __restrict__ tad,
                                               float* __restrict__ out) {
    __shared__ float red[256];
    int tid = threadIdx.x;
    float s = 0.0f;
    for (int i = tid; i < NPTS; i += 256) s += tad[i];
    red[tid] = s; __syncthreads();
    for (int st = 128; st > 0; st >>= 1) {
        if (tid < st) red[tid] += red[tid + st];
        __syncthreads();
    }
    float losssum = red[0];
    __syncthreads();

    float* em = out + 3 + (size_t)ZSIZE;     // e_mean region (held counts)
    float ent = 0.0f;
    const float invN = 1.0f / (float)NPTS;
    for (int i = tid; i < NE; i += 256) {
        float e = em[i] * invN;
        em[i] = e;
        ent -= e * logf(e + 1e-10f);
    }
    red[tid] = ent; __syncthreads();
    for (int st = 128; st > 0; st >>= 1) {
        if (tid < st) red[tid] += red[tid + st];
        __syncthreads();
    }
    if (tid == 0) {
        float entropy = red[0];
        out[0]                 = 1.25f * losssum * (1.0f / (float)NPTS); // (1+BETA)*mean
        out[1 + (size_t)ZSIZE] = expf(entropy);   // perplexity
        out[2 + (size_t)ZSIZE] = entropy;         // diversity_loss
    }
}

// ---------------------------------------------------------------- launch
extern "C" void kernel_launch(void* const* d_in, const int* in_sizes, int n_in,
                              void* d_out, int out_size, void* d_ws, size_t ws_size,
                              hipStream_t stream) {
    const float* u    = (const float*)d_in[0];   // (16,64,64,129)
    const float* rcen = (const float*)d_in[1];   // (16,)
    const float* aw   = (const float*)d_in[2];   // (1024,128)
    float* out = (float*)d_out;
    float* ws  = (float*)d_ws;

    // workspace layout (floats):
    // invn[N] | topr[N] | top5[5N] | tad[N] | awh (131072 f16 = 65536 f32 slots)
    float*     invn = ws;
    int*       topr = (int*)(ws + NPTS);
    int*       top5 = (int*)(ws + 2 * NPTS);
    float*     tad  = ws + 7 * NPTS;
    _Float16*  awh  = (_Float16*)(ws + 8 * NPTS);
    float*     counts = out + 3 + (size_t)ZSIZE;

    k_init  <<<(NE + 255) / 256, 256, 0, stream>>>(counts);
    k_cvt_aw<<<(ABINS * SDIM + 255) / 256, 256, 0, stream>>>(aw, awh);
    k_prep  <<<(NPTS + 255) / 256, 256, 0, stream>>>(u, rcen, invn, topr);
    k_sim   <<<NPTS / (PTS_PER_WAVE * WAVES_PER_BLK), 256, 0, stream>>>(u, awh, invn, top5);
    k_assign<<<NPTS / 8, 256, 0, stream>>>(u, aw, rcen, topr, top5, out, tad, counts);
    k_final <<<1, 256, 0, stream>>>(tad, out);
}
